// PropagationGraphConvLayer_nBatch_geometric_78950088835522
// MI455X (gfx1250) — matrix operand order repaired and checked by reading, compiled, not run on hardware
//
#include <hip/hip_runtime.h>
#include <stdint.h>

// ---------------------------------------------------------------------------
// Types for CDNA5 WMMA (wave32): v_wmma_f32_16x16x32_bf16
// ---------------------------------------------------------------------------
typedef __attribute__((ext_vector_type(16))) __bf16 bf16x16;
typedef __attribute__((ext_vector_type(8)))  float  f32x8;

__device__ __forceinline__ float lrelu(float v) {
    return v >= 0.0f ? v : 0.01f * v;
}

// ---------------------------------------------------------------------------
// rootnode[i] = (int)root_idx[batch[i]]
// ---------------------------------------------------------------------------
__global__ void build_rootnode_kernel(int* __restrict__ rootnode,
                                      const long long* __restrict__ batch,
                                      const long long* __restrict__ root_idx,
                                      int N) {
    int i = blockIdx.x * blockDim.x + threadIdx.x;
    if (i < N) rootnode[i] = (int)root_idx[(int)batch[i]];
}

// ---------------------------------------------------------------------------
// Zero fill (graph-capture safe; no hipMemsetAsync needed)
// ---------------------------------------------------------------------------
__global__ void zero_fill_kernel(float4* __restrict__ p, long long n4) {
    long long i = (long long)blockIdx.x * blockDim.x + threadIdx.x;
    if (i < n4) p[i] = make_float4(0.f, 0.f, 0.f, 0.f);
}

// ---------------------------------------------------------------------------
// Convert weight [K x 256] f32 row-major -> WMMA B-fragment-ordered hi/lo bf16.
// B (32x16, bf16) layout per lane l: n = l&15, h = l>>4,
//   element i (0..15) = B[kc*32 + (i<8 ? 8*h + i : 16 + 8*h + (i-8))][ntile*16 + n]
// stored at  ((kc*16 + ntile)*32 + lane)*16 + i
// ---------------------------------------------------------------------------
__global__ void convert_weights_kernel(const float* __restrict__ W, int K,
                                       uint16_t* __restrict__ hi,
                                       uint16_t* __restrict__ lo) {
    int idx = blockIdx.x * blockDim.x + threadIdx.x;
    if (idx >= K * 256) return;
    int n = idx & 255;
    int k = idx >> 8;
    float w = W[idx];
    __bf16 h  = (__bf16)w;
    float rem = w - (float)h;
    __bf16 l  = (__bf16)rem;
    int kc = k >> 5, within = k & 31;
    int lane = (((within >> 3) & 1) << 4) | (n & 15);
    int nt = n >> 4;
    int i  = (within & 7) | ((within >> 4) << 3);
    size_t o = (((size_t)kc * 16 + nt) * 32 + lane) * 16 + i;
    hi[o] = __builtin_bit_cast(uint16_t, h);
    lo[o] = __builtin_bit_cast(uint16_t, l);
}

// ---------------------------------------------------------------------------
// Edge scatter:  out[dst[e], :] += values[e] * src[src[e], :]   (256 cols)
// One thread per (edge, 4 columns) -> float4 load + 4 hw f32 atomics.
// ---------------------------------------------------------------------------
__global__ void scatter_edges_kernel(const float* __restrict__ src,
                                     float* __restrict__ out,
                                     const long long* __restrict__ adjs,
                                     const float* __restrict__ vals, int E) {
    long long tid = (long long)blockIdx.x * blockDim.x + threadIdx.x;
    if (tid >= (long long)E * 64) return;
    int e = (int)(tid >> 6);
    int c = ((int)tid & 63) << 2;
    int s = (int)adjs[e];
    int d = (int)adjs[(size_t)E + e];
    float wv = vals[e];
    const float4 v = *(const float4*)(src + (size_t)s * 256 + c);
    float* o = out + (size_t)d * 256 + c;
    unsafeAtomicAdd(o + 0, wv * v.x);
    unsafeAtomicAdd(o + 1, wv * v.y);
    unsafeAtomicAdd(o + 2, wv * v.z);
    unsafeAtomicAdd(o + 3, wv * v.w);
}

// ---------------------------------------------------------------------------
// H[i, c] += b[c]   (256 cols)
// ---------------------------------------------------------------------------
__global__ void add_bias_kernel(float* __restrict__ H, const float* __restrict__ b,
                                long long total) {
    long long i = (long long)blockIdx.x * blockDim.x + threadIdx.x;
    if (i < total) H[i] += b[(int)(i & 255)];
}

// ---------------------------------------------------------------------------
// Build one 16x32 bf16 A-fragment pair (hi/lo split) from a row pointer.
// 16-bit A layout, element j of lane (half = lane>>4):
//   j<8 : K = kloc + 8*half + j ; j>=8 : K = kloc + 16 + 8*half + (j-8)
// ---------------------------------------------------------------------------
__device__ __forceinline__ void load_a_frag(const float* __restrict__ rowp,
                                            int kloc, int half,
                                            const float* __restrict__ bvec,
                                            int lky,
                                            bf16x16& ahi, bf16x16& alo) {
    float av[16];
    const float4 g0 = *(const float4*)(rowp + kloc + 8 * half);
    const float4 g1 = *(const float4*)(rowp + kloc + 8 * half + 4);
    const float4 g2 = *(const float4*)(rowp + kloc + 16 + 8 * half);
    const float4 g3 = *(const float4*)(rowp + kloc + 16 + 8 * half + 4);
    av[0] = g0.x;  av[1] = g0.y;  av[2]  = g0.z;  av[3]  = g0.w;
    av[4] = g1.x;  av[5] = g1.y;  av[6]  = g1.z;  av[7]  = g1.w;
    av[8] = g2.x;  av[9] = g2.y;  av[10] = g2.z;  av[11] = g2.w;
    av[12] = g3.x; av[13] = g3.y; av[14] = g3.z;  av[15] = g3.w;
    if (bvec) {
        const int kb0 = kloc + 8 * half;
        const int kb1 = kloc + 16 + 8 * half;
#pragma unroll
        for (int j = 0; j < 8; ++j) av[j] += bvec[kb0 + j];
#pragma unroll
        for (int j = 0; j < 8; ++j) av[8 + j] += bvec[kb1 + j];
    }
    if (lky) {
#pragma unroll
        for (int j = 0; j < 16; ++j) av[j] = lrelu(av[j]);
    }
#pragma unroll
    for (int j = 0; j < 16; ++j) {
        __bf16 h = (__bf16)av[j];
        ahi[j] = h;
        alo[j] = (__bf16)(av[j] - (float)h);
    }
}

// ---------------------------------------------------------------------------
// WMMA GEMM with virtual-concat A operand:
//   A(row, k) = k <  K1 : t1( A1[row, k] )            t1 = (+bias1[k])?, leaky?
//              k >= K1 : t2( A2[rowmap[row], k-K1] )  t2 = leaky?
//   Out[row, 0..255] = epi( A @ Bfrag )               epi = (+biasOut)?, leaky?
// B supplied as hi/lo fragment-ordered bf16; D = Ahi*Bhi + Ahi*Blo + Alo*Bhi
// (split-precision: ~f32 accuracy on the bf16 WMMA pipe).
// One wave -> 32 rows x 256 cols (2 row-tiles x 16 n-tiles = 32 accumulators);
// every B fragment load feeds 6 WMMAs. 4 waves / block.
// ---------------------------------------------------------------------------
__global__ __launch_bounds__(128)
void gemm_wmma_kernel(const float* __restrict__ A1, int K1,
                      const float* __restrict__ bias1, int leaky1,
                      const float* __restrict__ A2, int K2,
                      const int* __restrict__ rowmap, int leaky2,
                      const uint16_t* __restrict__ Bhi,
                      const uint16_t* __restrict__ Blo,
                      const float* __restrict__ biasOut, int leakyOut,
                      float* __restrict__ Out, int M, int Ktot) {
    const int lane  = threadIdx.x & 31;
    const int wave  = threadIdx.x >> 5;
    const int tile  = blockIdx.x * 4 + wave;   // 32-row tile
    const int mbase = tile * 32;
    const int half  = lane >> 4;               // 0 or 1
    const int mrow0 = min(mbase + (lane & 15), M - 1);       // row-tile 0 loads
    const int mrow1 = min(mbase + 16 + (lane & 15), M - 1);  // row-tile 1 loads

    int rr0 = 0, rr1 = 0;
    if (A2) {
        rr0 = rowmap[mrow0];
        rr1 = rowmap[mrow1];
    }

    f32x8 acc0[16], acc1[16];
#pragma unroll
    for (int t = 0; t < 16; ++t) {
        f32x8 z = {0.f, 0.f, 0.f, 0.f, 0.f, 0.f, 0.f, 0.f};
        acc0[t] = z;
        acc1[t] = z;
    }

    for (int kc = 0; kc < Ktot; kc += 32) {
        const float* rowp0;
        const float* rowp1;
        const float* bvec;
        int kloc, lky;
        if (kc < K1) {
            rowp0 = A1 + (size_t)mrow0 * K1;
            rowp1 = A1 + (size_t)mrow1 * K1;
            kloc  = kc;
            bvec  = bias1;
            lky   = leaky1;
        } else {
            rowp0 = A2 + (size_t)rr0 * K2;
            rowp1 = A2 + (size_t)rr1 * K2;
            kloc  = kc - K1;
            bvec  = nullptr;
            lky   = leaky2;
        }

        bf16x16 ahi0, alo0, ahi1, alo1;
        load_a_frag(rowp0, kloc, half, bvec, lky, ahi0, alo0);
        load_a_frag(rowp1, kloc, half, bvec, lky, ahi1, alo1);

        const size_t fb = (((size_t)(kc >> 5) * 16) * 32 + lane) * 16;
#pragma unroll
        for (int nt = 0; nt < 16; ++nt) {
            const bf16x16 bh = *(const bf16x16*)(Bhi + fb + (size_t)nt * 512);
            const bf16x16 bl = *(const bf16x16*)(Blo + fb + (size_t)nt * 512);
            acc0[nt] = __builtin_amdgcn_wmma_f32_16x16x32_bf16(
                false, ahi0, false, bh, (short)0, acc0[nt], false, false);
            acc1[nt] = __builtin_amdgcn_wmma_f32_16x16x32_bf16(
                false, ahi1, false, bh, (short)0, acc1[nt], false, false);
            acc0[nt] = __builtin_amdgcn_wmma_f32_16x16x32_bf16(
                false, ahi0, false, bl, (short)0, acc0[nt], false, false);
            acc1[nt] = __builtin_amdgcn_wmma_f32_16x16x32_bf16(
                false, ahi1, false, bl, (short)0, acc1[nt], false, false);
            acc0[nt] = __builtin_amdgcn_wmma_f32_16x16x32_bf16(
                false, alo0, false, bh, (short)0, acc0[nt], false, false);
            acc1[nt] = __builtin_amdgcn_wmma_f32_16x16x32_bf16(
                false, alo1, false, bh, (short)0, acc1[nt], false, false);
        }
    }

    // C/D layout: col n = lane&15, row = r + 8*half (per 16x16 tile)
    const int colbase = lane & 15;
#pragma unroll
    for (int nt = 0; nt < 16; ++nt) {
#pragma unroll
        for (int r = 0; r < 8; ++r) {
            const int col = nt * 16 + colbase;
            float bo = biasOut ? biasOut[col] : 0.0f;

            int mr0 = mbase + r + 8 * half;
            if (mr0 < M) {
                float v = acc0[nt][r] + bo;
                if (leakyOut) v = lrelu(v);
                Out[(size_t)mr0 * 256 + col] = v;
            }
            int mr1 = mbase + 16 + r + 8 * half;
            if (mr1 < M) {
                float v = acc1[nt][r] + bo;
                if (leakyOut) v = lrelu(v);
                Out[(size_t)mr1 * 256 + col] = v;
            }
        }
    }
}

// ---------------------------------------------------------------------------
// Host orchestration
// ---------------------------------------------------------------------------
extern "C" void kernel_launch(void* const* d_in, const int* in_sizes, int n_in,
                              void* d_out, int out_size, void* d_ws, size_t ws_size,
                              hipStream_t stream) {
    const float*     features = (const float*)d_in[0];
    const float*     values   = (const float*)d_in[1];
    const float*     W1       = (const float*)d_in[2];
    const float*     b1       = (const float*)d_in[3];
    const float*     W2       = (const float*)d_in[4];
    const float*     b2       = (const float*)d_in[5];
    const float*     Wl       = (const float*)d_in[6];
    const float*     bl       = (const float*)d_in[7];
    const long long* adjs     = (const long long*)d_in[8];
    const long long* batch    = (const long long*)d_in[9];
    const long long* root_idx = (const long long*)d_in[10];

    const int N = in_sizes[0] / 256;   // 100000
    const int E = in_sizes[1];         // 300000

    // ---- workspace carve-out (256B aligned) ----
    char* wp = (char*)d_ws;
    auto carve = [&](size_t bytes) -> char* {
        char* p = wp;
        wp += (bytes + 255) & ~(size_t)255;
        return p;
    };
    const size_t matBytes = (size_t)N * 256 * sizeof(float);
    float*    hpre  = (float*)carve(matBytes);              // GEMM1 out / GEMM2 out
    float*    H     = (float*)carve(matBytes);              // agg1 + b1
    float*    agg2  = (float*)carve(matBytes);              // agg of conv2
    uint16_t* W1hi  = (uint16_t*)carve((size_t)256 * 256 * 2);
    uint16_t* W1lo  = (uint16_t*)carve((size_t)256 * 256 * 2);
    uint16_t* W2hi  = (uint16_t*)carve((size_t)512 * 256 * 2);
    uint16_t* W2lo  = (uint16_t*)carve((size_t)512 * 256 * 2);
    uint16_t* Wlhi  = (uint16_t*)carve((size_t)512 * 256 * 2);
    uint16_t* Wllo  = (uint16_t*)carve((size_t)512 * 256 * 2);
    int*      rootn = (int*)carve((size_t)N * sizeof(int));

    const int numTiles   = (N + 31) / 32;       // 3125 (exact)
    const int gemmBlocks = (numTiles + 3) / 4;  // 4 waves / block
    const long long scatterThreads = (long long)E * 64;
    const int scatterBlocks = (int)((scatterThreads + 255) / 256);
    const long long n4 = (long long)N * 256 / 4;
    const int zeroBlocks = (int)((n4 + 255) / 256);

    // rootnode map + weight fragment conversion
    build_rootnode_kernel<<<(N + 255) / 256, 256, 0, stream>>>(rootn, batch, root_idx, N);
    convert_weights_kernel<<<(256 * 256 + 255) / 256, 256, 0, stream>>>(W1, 256, W1hi, W1lo);
    convert_weights_kernel<<<(512 * 256 + 255) / 256, 256, 0, stream>>>(W2, 512, W2hi, W2lo);
    convert_weights_kernel<<<(512 * 256 + 255) / 256, 256, 0, stream>>>(Wl, 512, Wlhi, Wllo);

    // conv1: hpre = features @ W1
    gemm_wmma_kernel<<<gemmBlocks, 128, 0, stream>>>(
        features, 256, nullptr, 0,
        nullptr, 0, nullptr, 0,
        W1hi, W1lo, nullptr, 0, hpre, N, 256);

    // H = segment_sum(w * hpre[src] -> dst) + b1
    zero_fill_kernel<<<zeroBlocks, 256, 0, stream>>>((float4*)H, n4);
    scatter_edges_kernel<<<scatterBlocks, 256, 0, stream>>>(hpre, H, adjs, values, E);
    add_bias_kernel<<<(int)(((long long)N * 256 + 255) / 256), 256, 0, stream>>>(
        H, b1, (long long)N * 256);

    // conv2: x2pre = leaky(concat(H, features[root])) @ W2  (virtual concat)
    gemm_wmma_kernel<<<gemmBlocks, 128, 0, stream>>>(
        H, 256, nullptr, 1,
        features, 256, rootn, 1,
        W2hi, W2lo, nullptr, 0, hpre, N, 512);

    // agg2 = segment_sum(w * x2pre[src] -> dst)
    zero_fill_kernel<<<zeroBlocks, 256, 0, stream>>>((float4*)agg2, n4);
    scatter_edges_kernel<<<scatterBlocks, 256, 0, stream>>>(hpre, agg2, adjs, values, E);

    // out = leaky( concat(leaky(agg2 + b2), H[root]) @ Wl + bl )
    gemm_wmma_kernel<<<gemmBlocks, 128, 0, stream>>>(
        agg2, 256, b2, 1,
        H, 256, rootn, 0,
        Wlhi, Wllo, bl, 1, (float*)d_out, N, 512);
}
